// MixerGatedDeltaNet_4612794876169
// MI455X (gfx1250) — matrix operand order repaired and checked by reading, not compiled, not run on hardware
//
#include <hip/hip_runtime.h>
#include <hip/hip_bf16.h>
#include <stddef.h>
#include <stdint.h>

#define D_MODEL   2048
#define TP        12320
#define NH        16
#define DH        128
#define HV        256
#define KEY_DIM   2048
#define VAL_DIM   4096
#define SEQ_L     2048
#define BATCH     2
#define MTOT      (BATCH * SEQ_L)   // 4096 rows in the GEMM
#define CH        32                // recurrence chunk length
#define NCH       (SEQ_L / CH)      // 64 sequential chunks

#define USE_ASYNC_LDS 1

typedef __attribute__((ext_vector_type(16))) __bf16        bf16x16;
typedef __attribute__((ext_vector_type(8)))  float         f32x8;
typedef __attribute__((ext_vector_type(4)))  float         f32x4;
typedef __attribute__((ext_vector_type(4)))  unsigned int  u32x4;
typedef __attribute__((ext_vector_type(2)))  unsigned int  u32x2;

union Frag16 { bf16x16 v; u32x4 q[2]; };

static __device__ inline unsigned short f2bf(float f) {
  // round-to-nearest-even f32 -> bf16 bits
  unsigned u = __builtin_bit_cast(unsigned, f);
  unsigned r = u + 0x7FFFu + ((u >> 16) & 1u);
  return (unsigned short)(r >> 16);
}

static __device__ inline float dev_silu(float x) { return x / (1.f + __expf(-x)); }

#if USE_ASYNC_LDS
// CDNA5 async global->LDS copies, tracked by ASYNCcnt (complete in order).
static __device__ inline void async_b128(unsigned lds_byte, const void* g) {
  asm volatile("global_load_async_to_lds_b128 %0, %1, off"
               :: "v"(lds_byte), "v"((unsigned long long)(uintptr_t)g)
               : "memory");
}
static __device__ inline void async_b64(unsigned lds_byte, const void* g) {
  asm volatile("global_load_async_to_lds_b64 %0, %1, off"
               :: "v"(lds_byte), "v"((unsigned long long)(uintptr_t)g)
               : "memory");
}
static __device__ inline void wait_async0()   { asm volatile("s_wait_asynccnt 0" ::: "memory"); }
static __device__ inline void wait_async_le5(){ asm volatile("s_wait_asynccnt 5" ::: "memory"); }
#endif
// generic-AS shared pointer: low 32 bits are the LDS byte address
static __device__ inline unsigned lds_addr(const void* p) {
  return (unsigned)(unsigned long long)(uintptr_t)p;
}

// ---------------------------------------------------------------------------
// Kernel 0a: one-time f32 -> bf16 convert (row-major copy) for X.
// ---------------------------------------------------------------------------
__global__ __launch_bounds__(256) void cvt_bf16(const float* __restrict__ src,
                                                unsigned short* __restrict__ dst,
                                                int n4) {
  int i = blockIdx.x * 256 + threadIdx.x;
  if (i < n4) {
    f32x4 v = *(const f32x4*)(src + (size_t)i * 4);
    u32x2 t;
    t.x = (unsigned)f2bf(v[0]) | ((unsigned)f2bf(v[1]) << 16);
    t.y = (unsigned)f2bf(v[2]) | ((unsigned)f2bf(v[3]) << 16);
    *(u32x2*)(dst + (size_t)i * 4) = t;
  }
}

// ---------------------------------------------------------------------------
// Kernel 0b: one-time transpose + convert: W (2048 x 12320, f32, [k][n])
//            -> WbT (12320 x 2048, bf16, [n][k]).  LDS-tiled 32x32.
// ---------------------------------------------------------------------------
__global__ __launch_bounds__(256) void cvt_w_transpose(
    const float* __restrict__ W, unsigned short* __restrict__ WT) {
  __shared__ float tile[32][33];
  const int n0 = blockIdx.x * 32;    // 385
  const int k0 = blockIdx.y * 32;    // 64
  const int tid = threadIdx.x;
  #pragma unroll
  for (int i = 0; i < 4; ++i) {
    int idx = tid + i * 256;
    int r = idx >> 5, c = idx & 31;                    // r: k, c: n (coalesced)
    tile[r][c] = W[(size_t)(k0 + r) * TP + (n0 + c)];
  }
  __syncthreads();
  #pragma unroll
  for (int i = 0; i < 4; ++i) {
    int idx = tid + i * 256;
    int r = idx >> 5, c = idx & 31;                    // r: n, c: k (coalesced)
    WT[(size_t)(n0 + r) * D_MODEL + (k0 + c)] = f2bf(tile[c][r]);
  }
}

// ---------------------------------------------------------------------------
// Tile staging for the GEMM: pure byte copy (A [m][k], B [n][k], both bf16,
// k-contiguous).  Uniform 5 async ops per thread per tile -> constant-imm
// s_wait_asynccnt double buffering.
// ---------------------------------------------------------------------------
static __device__ inline void stage_tile(
    const unsigned short* __restrict__ Xb, const unsigned short* __restrict__ Wb,
    unsigned short* Asd, unsigned short* Bsd,
    int m0, int n0, int k0, int tid) {
#if USE_ASYNC_LDS
  const unsigned ab = lds_addr(Asd), bbo = lds_addr(Bsd);
  #pragma unroll
  for (int i = 0; i < 2; ++i) {                       // A: 128x32 = 512 x 16B
    int u = tid + i * 256;
    async_b128(ab + u * 16, Xb + (size_t)(m0 + (u >> 2)) * D_MODEL + k0 + (u & 3) * 8);
  }
  #pragma unroll
  for (int i = 0; i < 2; ++i) {                       // B rows 0..127: 512 x 16B
    int u = tid + i * 256;
    async_b128(bbo + u * 16, Wb + (size_t)(n0 + (u >> 2)) * D_MODEL + k0 + (u & 3) * 8);
  }
  // B rows 128..159: 256 x 8B
  async_b64(bbo + 8192 + tid * 8,
            Wb + (size_t)(n0 + 128 + (tid >> 3)) * D_MODEL + k0 + (tid & 7) * 4);
#else
  #pragma unroll
  for (int i = 0; i < 2; ++i) {
    int u = tid + i * 256;
    *(u32x4*)(Asd + u * 8) = *(const u32x4*)(Xb + (size_t)(m0 + (u >> 2)) * D_MODEL + k0 + (u & 3) * 8);
    *(u32x4*)(Bsd + u * 8) = *(const u32x4*)(Wb + (size_t)(n0 + (u >> 2)) * D_MODEL + k0 + (u & 3) * 8);
  }
  *(u32x2*)(Bsd + 4096 + tid * 4) =
      *(const u32x2*)(Wb + (size_t)(n0 + 128 + (tid >> 3)) * D_MODEL + k0 + (tid & 7) * 4);
#endif
}

// ---------------------------------------------------------------------------
// Kernel 1: proj = Xb(4096x2048) @ WbT^T (-> 4096x12320), bf16 WMMA, f32 acc.
// Block tile 128(M) x 160(N), K-step 32, double-buffered async LDS staging.
// 8 waves = 4(M) x 2(N); each wave owns 2x5 WMMA 16x16 accumulators.
// ---------------------------------------------------------------------------
__global__ __launch_bounds__(256) void gdn_gemm_proj(
    const unsigned short* __restrict__ Xb, const unsigned short* __restrict__ Wb,
    float* __restrict__ P) {
  __shared__ __align__(16) unsigned short As[2][128 * 32];   // [m][k] bf16
  __shared__ __align__(16) unsigned short Bs[2][160 * 32];   // [n][k] bf16

  const int tid   = threadIdx.x;
  const int lane  = tid & 31;
  const int wave  = tid >> 5;
  const int wm    = wave >> 1;
  const int wn    = wave & 1;
  const int m0    = blockIdx.y * 128;
  const int n0    = blockIdx.x * 160;
  const int lr    = lane & 15;
  const int khalf = lane >> 4;

  f32x8 acc[2][5] = {};

  stage_tile(Xb, Wb, &As[0][0], &Bs[0][0], m0, n0, 0, tid);

  for (int it = 0; it < 64; ++it) {
    const int buf = it & 1;
    if (it + 1 < 64)
      stage_tile(Xb, Wb, &As[1 - buf][0], &Bs[1 - buf][0], m0, n0, (it + 1) * 32, tid);
#if USE_ASYNC_LDS
    if (it + 1 < 64) wait_async_le5(); else wait_async0();
#endif
    __syncthreads();

    const unsigned short* Asb = &As[buf][0];
    const unsigned short* Bsb = &Bs[buf][0];
    Frag16 a[2];
    #pragma unroll
    for (int mt = 0; mt < 2; ++mt) {
      int row = wm * 32 + mt * 16 + lr;
      a[mt].q[0] = *(const u32x4*)&Asb[row * 32 + khalf * 8];
      a[mt].q[1] = *(const u32x4*)&Asb[row * 32 + 16 + khalf * 8];
    }
    #pragma unroll
    for (int nt = 0; nt < 5; ++nt) {
      int col = wn * 80 + nt * 16 + lr;
      Frag16 bfr;
      bfr.q[0] = *(const u32x4*)&Bsb[col * 32 + khalf * 8];
      bfr.q[1] = *(const u32x4*)&Bsb[col * 32 + 16 + khalf * 8];
      #pragma unroll
      for (int mt = 0; mt < 2; ++mt) {
        acc[mt][nt] = __builtin_amdgcn_wmma_f32_16x16x32_bf16(
            false, a[mt].v, false, bfr.v, (short)0, acc[mt][nt], false, false);
      }
    }
    __syncthreads();
  }

  #pragma unroll
  for (int mt = 0; mt < 2; ++mt) {
    #pragma unroll
    for (int nt = 0; nt < 5; ++nt) {
      int col = n0 + wn * 80 + nt * 16 + lr;
      #pragma unroll
      for (int j = 0; j < 8; ++j) {
        int row = m0 + wm * 32 + mt * 16 + khalf * 8 + j;
        P[(size_t)row * TP + col] = acc[mt][nt][j];
      }
    }
  }
}

// ---------------------------------------------------------------------------
// Kernel 2: depthwise causal conv (4-tap) + SiLU, per-head L2-norm of q/k,
// beta = sigmoid(b), g = -exp(A_log)*softplus(A+dt_bias). One block per (b,l).
// ---------------------------------------------------------------------------
__global__ __launch_bounds__(256) void gdn_conv_norm(
    const float* __restrict__ P, const float* __restrict__ Wc,
    const float* __restrict__ A_log, const float* __restrict__ dt_bias,
    float* __restrict__ qn, float* __restrict__ kn, float* __restrict__ vb,
    float* __restrict__ gb, float* __restrict__ bb) {
  __shared__ float buf[4096];
  __shared__ float rn[32];

  const int bl  = blockIdx.x;
  const int l   = bl & (SEQ_L - 1);
  const int tid = threadIdx.x;
  const size_t rowb = (size_t)bl * TP;

  #pragma unroll
  for (int i = 0; i < 16; ++i) {
    int c = tid + i * 256;
    float acc = 0.f;
    #pragma unroll
    for (int j = 0; j < 4; ++j) {
      int ll = l - 3 + j;
      if (ll >= 0) acc += P[(size_t)(bl - 3 + j) * TP + c] * Wc[c * 4 + j];
    }
    buf[c] = dev_silu(acc);
  }
  __syncthreads();
  if (tid < 32) {
    float s = 0.f;
    for (int d = 0; d < DH; ++d) { float x = buf[tid * DH + d]; s += x * x; }
    rn[tid] = rsqrtf(s + 1e-6f);
  }
  __syncthreads();
  #pragma unroll
  for (int i = 0; i < 16; ++i) {
    int c = tid + i * 256;
    float y = buf[c] * rn[c >> 7];
    if (c < KEY_DIM) qn[(size_t)bl * KEY_DIM + c] = y;
    else             kn[(size_t)bl * KEY_DIM + (c - KEY_DIM)] = y;
  }
  #pragma unroll
  for (int i = 0; i < 16; ++i) {
    int c = 4096 + tid + i * 256;
    float acc = 0.f;
    #pragma unroll
    for (int j = 0; j < 4; ++j) {
      int ll = l - 3 + j;
      if (ll >= 0) acc += P[(size_t)(bl - 3 + j) * TP + c] * Wc[c * 4 + j];
    }
    vb[(size_t)bl * VAL_DIM + (c - 4096)] = dev_silu(acc);
  }
  if (tid < NH) {
    float bcol = P[rowb + 12288 + tid];
    float acol = P[rowb + 12304 + tid];
    float sp   = log1pf(__expf(acol + dt_bias[tid]));
    gb[(size_t)bl * NH + tid] = -__expf(A_log[tid]) * sp;
    bb[(size_t)bl * NH + tid] = 1.f / (1.f + __expf(-bcol));
  }
}

// ---------------------------------------------------------------------------
// Kernel 3: chunked gated delta-rule recurrence (C=32). One block per (b,h).
// State S (128x256 f32) lives entirely in WMMA accumulator registers.
// ---------------------------------------------------------------------------
__global__ __launch_bounds__(256) void gdn_recurrence_chunked(
    const float* __restrict__ qn, const float* __restrict__ kn,
    const float* __restrict__ vb, const float* __restrict__ gb,
    const float* __restrict__ bb, const float* __restrict__ P,
    float* __restrict__ out) {
  extern __shared__ __align__(16) char smem[];
  unsigned short* Sb   = (unsigned short*)smem;   // [v][d] 256x128 bf16 (S0 shadow)
  unsigned short* kqb  = Sb   + 256 * 128;        // [t][d] 64x128 bf16
  unsigned short* kpr  = kqb  + 64 * 128;         // [d][t] 128x32 bf16
  unsigned short* attb = kpr  + 128 * 32;         // [t][j] 32x32 bf16
  unsigned short* db16 = attb + 32 * 32;          // [v][t] 256x32 bf16
  float* k32  = (float*)(db16 + 256 * 32);        // [t][d]
  float* q32  = k32  + CH * DH;
  float* v32  = q32  + CH * DH;                   // [t][v]
  float* zbuf = v32  + CH * HV;                   // [t][v] 64x256
  float* ud   = zbuf + 2 * CH * HV;               // [t][v] 32x256
  float* pbuf = ud   + CH * HV;                   // [t][j] 64x32
  float* attf = pbuf + 2 * CH * CH;               // [t][j] 32x32
  float* gv   = attf + CH * CH;                   // G[32] eG[32] beta[32] eGct[32] eGC

  const int b     = blockIdx.x >> 4;
  const int h     = blockIdx.x & (NH - 1);
  const int tid   = threadIdx.x;
  const int lane  = tid & 31;
  const int wave  = tid >> 5;
  const int lr    = lane & 15;
  const int khalf = lane >> 4;
  const float scale = 0.0883883476483184f;  // 128^-0.5

  f32x8 SAcc[16] = {};

  for (int c = 0; c < NCH; ++c) {
    const size_t bl0 = (size_t)b * SEQ_L + (size_t)c * CH;

    // ---- phase 0: stage chunk k/q/v; scalars; publish bf16 shadow of S0 ----
#if USE_ASYNC_LDS
    {
      const unsigned ko = lds_addr(k32), qo = lds_addr(q32), vo = lds_addr(v32);
      #pragma unroll
      for (int i = 0; i < 4; ++i) {
        int u = tid + i * 256;
        int t = u >> 5, dp = u & 31;
        async_b128(ko + u * 16, kn + (bl0 + t) * KEY_DIM + h * DH + dp * 4);
        async_b128(qo + u * 16, qn + (bl0 + t) * KEY_DIM + h * DH + dp * 4);
      }
      #pragma unroll
      for (int i = 0; i < 8; ++i) {
        int u = tid + i * 256;
        int t = u >> 6, dp = u & 63;
        async_b128(vo + u * 16, vb + (bl0 + t) * VAL_DIM + h * HV + dp * 4);
      }
    }
#else
    #pragma unroll
    for (int i = 0; i < 4; ++i) {
      int u = tid + i * 256;
      int t = u >> 5, dp = u & 31;
      *(f32x4*)&k32[t * DH + dp * 4] = *(const f32x4*)&kn[(bl0 + t) * KEY_DIM + h * DH + dp * 4];
      *(f32x4*)&q32[t * DH + dp * 4] = *(const f32x4*)&qn[(bl0 + t) * KEY_DIM + h * DH + dp * 4];
    }
    #pragma unroll
    for (int i = 0; i < 8; ++i) {
      int u = tid + i * 256;
      int t = u >> 6, dp = u & 63;
      *(f32x4*)&v32[t * HV + dp * 4] = *(const f32x4*)&vb[(bl0 + t) * VAL_DIM + h * HV + dp * 4];
    }
#endif
    if (tid == 0) {
      float Ga = 0.f;
      for (int t = 0; t < CH; ++t) {
        Ga += gb[(bl0 + t) * NH + h];
        gv[t]      = Ga;
        gv[32 + t] = __expf(Ga);
        gv[64 + t] = bb[(bl0 + t) * NH + h];
      }
      for (int t = 0; t < CH; ++t) gv[96 + t] = __expf(Ga - gv[t]);
      gv[128] = __expf(Ga);
    }
    #pragma unroll
    for (int i = 0; i < 16; ++i) {
      int v = i * 16 + lr;
      #pragma unroll
      for (int j = 0; j < 8; ++j) {
        int d = wave * 16 + khalf * 8 + j;
        Sb[v * DH + d] = f2bf(SAcc[i][j]);
      }
    }
#if USE_ASYNC_LDS
    wait_async0();
#endif
    __syncthreads();

    // ---- phase 1: bf16 [K;Q] and decayed-k ----
    #pragma unroll
    for (int i = 0; i < 32; ++i) {
      int idx = tid + i * 256;
      int t = idx >> 7, d = idx & 127;
      float val = (t < CH) ? k32[t * DH + d] : q32[(t - CH) * DH + d];
      kqb[idx] = f2bf(val);
    }
    #pragma unroll
    for (int i = 0; i < 16; ++i) {
      int idx = tid + i * 256;
      int d = idx >> 5, t = idx & 31;
      kpr[idx] = f2bf(gv[96 + t] * k32[t * DH + d]);
    }
    __syncthreads();

    // ---- phase 2: Z = [K;Q].S0 (64x256,K=128); P = [K;Q].K^T (64x32) ----
    for (int ti = wave; ti < 64; ti += 8) {
      int mt = ti >> 4, nt = ti & 15;
      f32x8 accz = {};
      #pragma unroll
      for (int ks = 0; ks < 4; ++ks) {
        Frag16 a, bf;
        int ro = (mt * 16 + lr) * DH + ks * 32;
        a.q[0] = *(const u32x4*)&kqb[ro + khalf * 8];
        a.q[1] = *(const u32x4*)&kqb[ro + 16 + khalf * 8];
        int co = (nt * 16 + lr) * DH + ks * 32 + khalf * 8;
        bf.q[0] = *(const u32x4*)&Sb[co];
        bf.q[1] = *(const u32x4*)&Sb[co + 16];
        accz = __builtin_amdgcn_wmma_f32_16x16x32_bf16(
            false, a.v, false, bf.v, (short)0, accz, false, false);
      }
      int col = nt * 16 + lr;
      #pragma unroll
      for (int j = 0; j < 8; ++j)
        zbuf[(mt * 16 + khalf * 8 + j) * HV + col] = accz[j];
    }
    {
      int mt = wave >> 1, nt = wave & 1;
      f32x8 accp = {};
      #pragma unroll
      for (int ks = 0; ks < 4; ++ks) {
        Frag16 a, bf;
        int ro = (mt * 16 + lr) * DH + ks * 32;
        a.q[0] = *(const u32x4*)&kqb[ro + khalf * 8];
        a.q[1] = *(const u32x4*)&kqb[ro + 16 + khalf * 8];
        int co = (nt * 16 + lr) * DH + ks * 32 + khalf * 8;
        bf.q[0] = *(const u32x4*)&kqb[co];
        bf.q[1] = *(const u32x4*)&kqb[co + 16];
        accp = __builtin_amdgcn_wmma_f32_16x16x32_bf16(
            false, a.v, false, bf.v, (short)0, accp, false, false);
      }
      int col = nt * 16 + lr;
      #pragma unroll
      for (int j = 0; j < 8; ++j)
        pbuf[(mt * 16 + khalf * 8 + j) * CH + col] = accp[j];
    }
    __syncthreads();

    // ---- phase 3: u and attention matrices ----
    #pragma unroll
    for (int i = 0; i < 32; ++i) {
      int idx = tid + i * 256;
      int t = idx >> 8;
      ud[idx] = gv[64 + t] * (v32[idx] - gv[32 + t] * zbuf[idx]);
    }
    #pragma unroll
    for (int i = 0; i < 4; ++i) {
      int idx = tid + i * 256;
      int t = idx >> 5, j = idx & 31;
      float dec = __expf(gv[t] - gv[j]);
      attf[idx] = (j < t) ? gv[64 + t] * dec * pbuf[idx] : 0.f;
      float qk  = pbuf[(CH + t) * CH + j];
      attb[idx] = f2bf((j <= t) ? dec * qk : 0.f);
    }
    __syncthreads();

    // ---- phase 4: forward substitution (per v column), delta -> bf16 ----
    {
      int v = tid;
      for (int t = 0; t < CH; ++t) {
        float x = ud[t * HV + v];
        for (int j = 0; j < t; ++j) x -= attf[t * CH + j] * ud[j * HV + v];
        ud[t * HV + v] = x;
      }
      #pragma unroll
      for (int t = 0; t < CH; ++t) db16[v * CH + t] = f2bf(ud[t * HV + v]);
    }
    __syncthreads();

    // ---- phase 5: O = e^G * Z_q + B_att.delta ; fused SiLU gating ----
    #pragma unroll
    for (int i = 0; i < 4; ++i) {
      int ti = wave * 4 + i;
      int mt = ti >> 4, nt = ti & 15;
      int col = nt * 16 + lr;
      f32x8 acco;
      #pragma unroll
      for (int j = 0; j < 8; ++j) {
        int t = mt * 16 + khalf * 8 + j;
        acco[j] = gv[32 + t] * zbuf[(CH + t) * HV + col];
      }
      Frag16 a, bf;
      int ro = (mt * 16 + lr) * CH;
      a.q[0] = *(const u32x4*)&attb[ro + khalf * 8];
      a.q[1] = *(const u32x4*)&attb[ro + 16 + khalf * 8];
      int co = col * CH + khalf * 8;
      bf.q[0] = *(const u32x4*)&db16[co];
      bf.q[1] = *(const u32x4*)&db16[co + 16];
      acco = __builtin_amdgcn_wmma_f32_16x16x32_bf16(
          false, a.v, false, bf.v, (short)0, acco, false, false);
      #pragma unroll
      for (int j = 0; j < 8; ++j) {
        int t = mt * 16 + khalf * 8 + j;
        size_t bl = bl0 + t;
        float gate = P[bl * TP + 8192 + h * HV + col];
        out[(bl * NH + h) * HV + col] = scale * acco[j] * dev_silu(gate);
      }
    }

    // ---- phase 6: S <- e^{G_C} S0 + K'^T.delta (registers) ----
    {
      const float eGC = gv[128];
      Frag16 a;
      int ro = (wave * 16 + lr) * CH;
      a.q[0] = *(const u32x4*)&kpr[ro + khalf * 8];
      a.q[1] = *(const u32x4*)&kpr[ro + 16 + khalf * 8];
      #pragma unroll
      for (int i = 0; i < 16; ++i) {
        SAcc[i] = SAcc[i] * eGC;
        Frag16 bf;
        int co = (i * 16 + lr) * CH + khalf * 8;
        bf.q[0] = *(const u32x4*)&db16[co];
        bf.q[1] = *(const u32x4*)&db16[co + 16];
        SAcc[i] = __builtin_amdgcn_wmma_f32_16x16x32_bf16(
            false, a.v, false, bf.v, (short)0, SAcc[i], false, false);
      }
    }
    __syncthreads();
  }
}

// ---------------------------------------------------------------------------
extern "C" void kernel_launch(void* const* d_in, const int* in_sizes, int n_in,
                              void* d_out, int out_size, void* d_ws, size_t ws_size,
                              hipStream_t stream) {
  const float* X   = (const float*)d_in[0];   // hidden_states (2,2048,2048)
  const float* Wp  = (const float*)d_in[1];   // W_qkvgba (2048,12320)
  const float* Wc  = (const float*)d_in[2];   // W_conv (8192,4)
  const float* Al  = (const float*)d_in[3];   // A_log (16)
  const float* db  = (const float*)d_in[4];   // dt_bias (16)
  float* out = (float*)d_out;

  float* ws   = (float*)d_ws;
  float* proj = ws;                                   // 4096*12320 f32
  float* qn   = proj + (size_t)MTOT * TP;
  float* kn   = qn   + (size_t)MTOT * KEY_DIM;
  float* vb   = kn   + (size_t)MTOT * KEY_DIM;
  float* gb   = vb   + (size_t)MTOT * VAL_DIM;
  float* bb   = gb   + (size_t)MTOT * NH;
  unsigned short* Xb = (unsigned short*)(bb + (size_t)MTOT * NH);  // 4096x2048 bf16
  unsigned short* Wb = Xb + (size_t)MTOT * D_MODEL;                // 12320x2048 bf16 ([n][k])

  // one-time bf16 conversion (+ W transpose) so the GEMM hot loop is pure copy
  int n4x = MTOT * D_MODEL / 4;
  cvt_bf16<<<(n4x + 255) / 256, 256, 0, stream>>>(X, Xb, n4x);
  cvt_w_transpose<<<dim3(TP / 32, D_MODEL / 32), 256, 0, stream>>>(Wp, Wb);

  dim3 gg(TP / 160, MTOT / 128);                       // 77 x 32
  gdn_gemm_proj<<<gg, 256, 0, stream>>>(Xb, Wb, proj);
  gdn_conv_norm<<<MTOT, 256, 0, stream>>>(proj, Wc, Al, db, qn, kn, vb, gb, bb);

  size_t smem3 = (size_t)((256*128 + 64*128 + 128*32 + 32*32 + 256*32) * 2
                        + (2*CH*DH + CH*HV + 2*CH*HV + CH*HV + 2*CH*CH + CH*CH + 136) * 4);
  gdn_recurrence_chunked<<<BATCH * NH, 256, smem3, stream>>>(qn, kn, vb, gb, bb, proj, out);
}
